// mfh_coatt_embed_ocr_12103217840659
// MI455X (gfx1250) — matrix operand order repaired and checked by reading, not compiled, not run on hardware
//
#include <hip/hip_runtime.h>
#include <hip/hip_bf16.h>

// ---------------- sizes from the reference ----------------
#define kB     32
#define kT     14
#define kH     1024
#define kJ     5000
#define kO     1000
#define kKF    5
#define kIMGC  2048
#define kNIMG  196
#define kTOK   50
#define kTOKE  300
#define kANS   3000
#define kOUTC  3051   // 1 + 3000 + 50

typedef __attribute__((ext_vector_type(16))) _Float16 v16h;
typedef __attribute__((ext_vector_type(8)))  _Float16 v8h;
typedef __attribute__((ext_vector_type(8)))  float    v8f;
typedef __attribute__((ext_vector_type(4)))  float    v4f;

#define GEMM_ACC  1
#define GEMM_RELU 2

// ============================================================
// Generic tiled WMMA GEMM:  C[m,n] (+)= A[m,k] * B[k,n] (+bias[m])
// A: row-major, lda (K and lda multiples of 4 -> true for every GEMM here).
// B: arbitrary strides. C: arbitrary strides. Batch via gridDim.z.
// Tile: 256(M) x 32(N) x 32(K), 8 waves, each wave -> 32x32 C
// (2 A-frags x 2 B-frags = 4 WMMAs per K-tile, B frags reused).
//
// Loads are UNCONDITIONAL from clamped addresses (no exec-mask predication,
// no select at load site): garbage in m/n padding is discarded by the
// epilogue bounds checks; only the K-tail is zeroed, at stage() time, and
// only on the last K-tile (uniform branch). Next tile's global loads are
// issued before the WMMAs so vmem overlaps matrix math.
// ============================================================
__global__ __launch_bounds__(256)
void wmma_gemm_f32(const float* __restrict__ A, long lda, long aBatch,
                   const float* __restrict__ Bm, long ldbK, long ldbN, long bBatch,
                   float* __restrict__ C, long csm, long csn, long cBatch,
                   const float* __restrict__ bias,
                   int M, int N, int K, int flags)
{
    __shared__ _Float16 sA[256 * 32];     // [row][k]  16KB
    __shared__ _Float16 sB[32 * 32];      // [n][k]    (transposed, contiguous frags)

    const int z = blockIdx.z;
    A  += (long)z * aBatch;
    Bm += (long)z * bBatch;
    C  += (long)z * cBatch;

    const int m0   = blockIdx.y * 256;
    const int n0   = blockIdx.x * 32;
    const int tid  = threadIdx.x;
    const int wave = tid >> 5;
    const int lane = tid & 31;
    const int half = lane >> 4;
    const int l16  = lane & 15;

    // ---- staging geometry ----
    // A: thread -> rows (tid>>1) and (tid>>1)+128, 16 consecutive k at
    // (tid&1)*16 each: 2 x (4 x float4). Rows clamped; garbage rows are
    // discarded by the epilogue m-bounds check.
    const int aRow0 = tid >> 1;            // 0..127
    const int aRow1 = aRow0 + 128;         // 128..255
    const int aKoff = (tid & 1) * 16;
    const int gm0   = m0 + aRow0;
    const int gm1   = m0 + aRow1;
    const float* aBase0 = A + (long)((gm0 < M) ? gm0 : (M - 1)) * lda;
    const float* aBase1 = A + (long)((gm1 < M) ? gm1 : (M - 1)) * lda;

    v4f   aR[8];
    float bR[4];

    auto loadA = [&](int k0) {
        #pragma unroll
        for (int i = 0; i < 4; ++i) {
            int gk  = k0 + aKoff + i * 4;
            int gkc = (gk < K) ? gk : 0;          // K%4==0 -> chunk all-or-none
            aR[i]     = *(const v4f*)(aBase0 + gkc);
            aR[4 + i] = *(const v4f*)(aBase1 + gkc);
        }
    };
    auto loadB = [&](int k0) {
        #pragma unroll
        for (int i = 0; i < 4; ++i) {
            int idx = tid + i * 256;              // 0..1023
            int kk  = idx >> 5, nn = idx & 31;
            int gk  = k0 + kk, gn = n0 + nn;
            int gkc = (gk < K) ? gk : (K - 1);
            int gnc = (gn < N) ? gn : (N - 1);
            bR[i] = Bm[(long)gkc * ldbK + (long)gnc * ldbN];  // unconditional
        }
    };
    auto stageRow = [&](const v4f* a4, int row) {
        v8h h0, h1;
        #pragma unroll
        for (int j = 0; j < 4; ++j) {
            h0[j]     = (_Float16)a4[0][j];
            h0[4 + j] = (_Float16)a4[1][j];
            h1[j]     = (_Float16)a4[2][j];
            h1[4 + j] = (_Float16)a4[3][j];
        }
        *(v8h*)&sA[row * 32 + aKoff]     = h0;    // ds_store_b128
        *(v8h*)&sA[row * 32 + aKoff + 8] = h1;    // ds_store_b128
    };
    auto stage = [&](int k0) {
        if (k0 + 32 <= K) {                       // fast path: no masking
            stageRow(&aR[0], aRow0);
            stageRow(&aR[4], aRow1);
            #pragma unroll
            for (int i = 0; i < 4; ++i) {
                int idx = tid + i * 256;
                int kk = idx >> 5, nn = idx & 31;
                sB[nn * 32 + kk] = (_Float16)bR[i];
            }
        } else {                                   // K tail: zero invalid k
            v4f av[8];
            #pragma unroll
            for (int i = 0; i < 4; ++i) {
                bool ok = (k0 + aKoff + i * 4) < K;
                av[i]     = ok ? aR[i]     : (v4f)0.0f;
                av[4 + i] = ok ? aR[4 + i] : (v4f)0.0f;
            }
            stageRow(&av[0], aRow0);
            stageRow(&av[4], aRow1);
            #pragma unroll
            for (int i = 0; i < 4; ++i) {
                int idx = tid + i * 256;
                int kk = idx >> 5, nn = idx & 31;
                float v = ((k0 + kk) < K) ? bR[i] : 0.0f;
                sB[nn * 32 + kk] = (_Float16)v;
            }
        }
    };

    v8f acc00 = {}, acc01 = {};   // A-frag 0 x B-frag {0,1}
    v8f acc10 = {}, acc11 = {};   // A-frag 1 x B-frag {0,1}

    const int kTiles = (K + 31) >> 5;
    loadA(0);
    loadB(0);

    for (int kt = 0; kt < kTiles; ++kt) {
        __syncthreads();                   // LDS free from previous compute
        stage(kt << 5);
        __syncthreads();

        // issue next tile's global loads before the WMMAs (overlap)
        if (kt + 1 < kTiles) {
            loadA((kt + 1) << 5);
            loadB((kt + 1) << 5);
        }
        if (kt + 2 < kTiles)
            __builtin_prefetch(aBase0 + ((kt + 2) << 5) + aKoff, 0, 1);

        // ---- A fragments (16x32 each) per CDNA5 ISA 7.12.2 ----
        v16h afrag0, afrag1;
        {
            int r0 = wave * 32 + l16;          // rows [wave*32, +16)
            int r1 = r0 + 16;                  // rows [wave*32+16, +16)
            v8h a0 = *(const v8h*)&sA[r0 * 32 + half * 8];
            v8h a1 = *(const v8h*)&sA[r0 * 32 + 16 + half * 8];
            v8h a2 = *(const v8h*)&sA[r1 * 32 + half * 8];
            v8h a3 = *(const v8h*)&sA[r1 * 32 + 16 + half * 8];
            #pragma unroll
            for (int i = 0; i < 8; ++i) {
                afrag0[i] = a0[i]; afrag0[i + 8] = a1[i];
                afrag1[i] = a2[i]; afrag1[i + 8] = a3[i];
            }
        }

        // ---- B fragments: two 16-col halves; each reused by both A frags ----
        {
            v8h b0 = *(const v8h*)&sB[(0 * 16 + l16) * 32 + half * 8];
            v8h b1 = *(const v8h*)&sB[(0 * 16 + l16) * 32 + 16 + half * 8];
            v16h bfrag;
            #pragma unroll
            for (int i = 0; i < 8; ++i) { bfrag[i] = b0[i]; bfrag[i + 8] = b1[i]; }
            acc00 = __builtin_amdgcn_wmma_f32_16x16x32_f16(
                false, afrag0, false, bfrag, (short)0, acc00, false, false);
            acc10 = __builtin_amdgcn_wmma_f32_16x16x32_f16(
                false, afrag1, false, bfrag, (short)0, acc10, false, false);
        }
        {
            v8h b0 = *(const v8h*)&sB[(1 * 16 + l16) * 32 + half * 8];
            v8h b1 = *(const v8h*)&sB[(1 * 16 + l16) * 32 + 16 + half * 8];
            v16h bfrag;
            #pragma unroll
            for (int i = 0; i < 8; ++i) { bfrag[i] = b0[i]; bfrag[i + 8] = b1[i]; }
            acc01 = __builtin_amdgcn_wmma_f32_16x16x32_f16(
                false, afrag0, false, bfrag, (short)0, acc01, false, false);
            acc11 = __builtin_amdgcn_wmma_f32_16x16x32_f16(
                false, afrag1, false, bfrag, (short)0, acc11, false, false);
        }
    }

    // ---- epilogue: C layout per ISA (VGPR v -> row half*8+v, col = l16) ----
    #pragma unroll
    for (int mi = 0; mi < 2; ++mi) {
        #pragma unroll
        for (int j = 0; j < 2; ++j) {
            v8f acc = mi ? (j ? acc11 : acc10) : (j ? acc01 : acc00);
            int n = n0 + j * 16 + l16;
            if (n >= N) continue;
            #pragma unroll
            for (int v = 0; v < 8; ++v) {
                int m = m0 + wave * 32 + mi * 16 + half * 8 + v;
                if (m >= M) continue;
                float r = acc[v];
                if (bias) r += bias[m];
                long o = (long)m * csm + (long)n * csn;
                if (flags & GEMM_ACC) r += C[o];
                if (flags & GEMM_RELU) r = fmaxf(r, 0.0f);
                C[o] = r;
            }
        }
    }
}

// ============================================================
// Small utility kernels
// ============================================================
__global__ void zero_k(float* p, long n) {
    long i = (long)blockIdx.x * 256 + threadIdx.x;
    if (i < n) p[i] = 0.0f;
}

__global__ void add2_k(float* d, const float* a, const float* b, int n) {
    int i = blockIdx.x * 256 + threadIdx.x;
    if (i < n) d[i] = a[i] + b[i];
}

// x_seq[t][b][0:600] = [tanh(emb[data[b,t]]), glove[b,t]]
__global__ void embed_k(const int* __restrict__ data, const float* __restrict__ emb,
                        const float* __restrict__ glove, float* __restrict__ x)
{
    long idx = (long)blockIdx.x * 256 + threadIdx.x;
    if (idx >= (long)kT * kB * 600) return;
    int c = (int)(idx % 600);
    long r = idx / 600;
    int b = (int)(r % kB);
    int t = (int)(r / kB);
    long o = ((long)t * kB + b) * 600 + c;
    if (c < 300) x[o] = tanhf(emb[(long)data[b * kT + t] * 300 + c]);
    else         x[o] = glove[((long)b * kT + t) * 300 + (c - 300)];
}

// gates g stored [4096][B]; torch order i,f,g,o
__global__ void lstm_cell_k(const float* __restrict__ g, float* __restrict__ h,
                            float* __restrict__ c, float* __restrict__ hs_t)
{
    int idx = blockIdx.x * 256 + threadIdx.x;
    if (idx >= kB * kH) return;
    int b = idx >> 10, hh = idx & 1023;
    float gi = g[(long)(hh)        * kB + b];
    float gf = g[(long)(kH + hh)   * kB + b];
    float gg = g[(long)(2*kH + hh) * kB + b];
    float go = g[(long)(3*kH + hh) * kB + b];
    float si = 1.0f / (1.0f + expf(-gi));
    float sf = 1.0f / (1.0f + expf(-gf));
    float so = 1.0f / (1.0f + expf(-go));
    float cn = sf * c[idx] + si * tanhf(gg);
    float hn = so * tanhf(cn);
    c[idx] = cn; h[idx] = hn; hs_t[idx] = hn;
}

// q-attention logits over n = t*B+b (GQ==1)
__global__ void qatt2_k(const float* __restrict__ a1, const float* __restrict__ W,
                        const float* __restrict__ bb, float* __restrict__ out)
{
    int n = blockIdx.x * 256 + threadIdx.x;
    if (n >= kT * kB) return;
    float s = bb[0];
    for (int c = 0; c < 512; ++c) s += W[c] * a1[(long)c * (kT * kB) + n];
    out[n] = s;
}

__global__ void q_softmax_k(float* qlog) {
    int b = threadIdx.x;
    if (b >= kB) return;
    float mx = -1e30f;
    for (int t = 0; t < kT; ++t) mx = fmaxf(mx, qlog[t * kB + b]);
    float e[kT], s = 0.0f;
    for (int t = 0; t < kT; ++t) { e[t] = expf(qlog[t * kB + b] - mx); s += e[t]; }
    for (int t = 0; t < kT; ++t) qlog[t * kB + b] = e[t] / s;
}

__global__ void qfeat_k(const float* __restrict__ attw, const float* __restrict__ hs,
                        float* __restrict__ qf)
{
    int idx = blockIdx.x * 256 + threadIdx.x;
    if (idx >= kB * kH) return;
    int b = idx >> 10, h = idx & 1023;
    float s = 0.0f;
    for (int t = 0; t < kT; ++t)
        s += attw[t * kB + b] * hs[(long)t * kB * kH + (long)b * kH + h];
    qf[idx] = s;
}

// MFB sum-pool over K_FAC: out[b][o][n] = sum_k qp[o*5+k][b] * vp[z][o*5+k][n]
__global__ void mfb_pool_k(const float* __restrict__ qp, const float* __restrict__ vp,
                           float* __restrict__ out, int b0, int chunk, int Nloc, long vpz)
{
    long idx = (long)blockIdx.x * 256 + threadIdx.x;
    long per = (long)kO * Nloc;
    if (idx >= (long)chunk * per) return;
    int z = (int)(idx / per);
    long rem = idx % per;
    int o = (int)(rem / Nloc), n = (int)(rem % Nloc);
    int b = b0 + z;
    const float* v = vp + (long)z * vpz;
    float s = 0.0f;
    #pragma unroll
    for (int k = 0; k < kKF; ++k) {
        int j = o * kKF + k;
        s += qp[(long)j * kB + b] * v[(long)j * Nloc + n];
    }
    out[((long)b * kO + o) * Nloc + n] = s;
}

// in-place signed-sqrt + per-b sum of squares (block reduce + atomic)
__global__ void ssqrt_norm_k(float* __restrict__ p, int perB, long bStride,
                             int slot, float* __restrict__ norms)
{
    __shared__ float red[256];
    int b = blockIdx.y;
    int i = blockIdx.x * 256 + threadIdx.x;
    float sq = 0.0f;
    if (i < perB) {
        float v = p[(long)b * bStride + i];
        float s = (v >= 0.0f) ? sqrtf(v) : -sqrtf(-v);
        p[(long)b * bStride + i] = s;
        sq = s * s;
    }
    red[threadIdx.x] = sq; __syncthreads();
    for (int s = 128; s > 0; s >>= 1) {
        if (threadIdx.x < s) red[threadIdx.x] += red[threadIdx.x + s];
        __syncthreads();
    }
    if (threadIdx.x == 0) atomicAdd(&norms[b * 8 + slot], red[0]);
}

__global__ void l2div_k(float* __restrict__ p, int perB, long bStride,
                        int slot, const float* __restrict__ norms)
{
    int b = blockIdx.y;
    int i = blockIdx.x * 256 + threadIdx.x;
    if (i >= perB) return;
    float nrm = fmaxf(sqrtf(norms[b * 8 + slot]), 1e-12f);
    p[(long)b * bStride + i] /= nrm;
}

// attention logits: out[b][g][n] = b2[g] + sum_c W[g][c]*a1[b][c][n]
__global__ void att2_k(const float* __restrict__ a1, long a1BStride,
                       const float* __restrict__ W, const float* __restrict__ b2,
                       float* __restrict__ out, int G, int Nloc)
{
    int idx = blockIdx.x * 256 + threadIdx.x;
    int total = kB * G * Nloc;
    if (idx >= total) return;
    int n = idx % Nloc;
    int g = (idx / Nloc) % G;
    int bb = idx / (Nloc * G);
    const float* a = a1 + (long)bb * a1BStride;
    float s = b2[g];
    for (int c = 0; c < 512; ++c) s += W[g * 512 + c] * a[(long)c * Nloc + n];
    out[idx] = s;
}

// softmax over Nloc; one block per (b,g) row (Nloc <= 256)
__global__ void softmax_loc_k(float* __restrict__ p, int Nloc) {
    __shared__ float red[256];
    float* row = p + (long)blockIdx.x * Nloc;
    int t = threadIdx.x;
    float v = (t < Nloc) ? row[t] : -1e30f;
    red[t] = v; __syncthreads();
    for (int s = 128; s > 0; s >>= 1) {
        if (t < s) red[t] = fmaxf(red[t], red[t + s]);
        __syncthreads();
    }
    float mx = red[0]; __syncthreads();
    float e = (t < Nloc) ? expf(v - mx) : 0.0f;
    red[t] = e; __syncthreads();
    for (int s = 128; s > 0; s >>= 1) {
        if (t < s) red[t] += red[t + s];
        __syncthreads();
    }
    if (t < Nloc) row[t] = e / red[0];
}

// out[b][g][c] = sum_n att[b][g][n] * V(b,c,n)
__global__ void att_wsum_k(const float* __restrict__ att, const float* __restrict__ V,
                           float* __restrict__ out, int G, int C, int Nloc,
                           long vbs, long vcs, long vns)
{
    long idx = (long)blockIdx.x * 256 + threadIdx.x;
    long total = (long)kB * G * C;
    if (idx >= total) return;
    int c = (int)(idx % C);
    int g = (int)((idx / C) % G);
    int b = (int)(idx / ((long)C * G));
    const float* a = att + ((long)b * G + g) * Nloc;
    const float* v = V + (long)b * vbs + (long)c * vcs;
    float s = 0.0f;
    for (int n = 0; n < Nloc; ++n) s += a[n] * v[(long)n * vns];
    out[idx] = s;
}

// MFH: z2=qpA*vpA (per j), z3=qpB*vpB*z2; pool K_FAC, signed-sqrt, accumulate norms
__global__ void mfh_pool_k(const float* __restrict__ qpA, const float* __restrict__ vpA,
                           const float* __restrict__ qpB, const float* __restrict__ vpB,
                           float* __restrict__ feat, int baseA, int baseB,
                           int slotA, int slotB, float* __restrict__ norms)
{
    __shared__ float red[256];
    int b = blockIdx.y;
    int o = blockIdx.x * 256 + threadIdx.x;
    float s2 = 0.0f, s3 = 0.0f;
    if (o < kO) {
        #pragma unroll
        for (int k = 0; k < kKF; ++k) {
            int j = o * kKF + k;
            float z2 = qpA[(long)j * kB + b] * vpA[(long)j * kB + b];
            s2 += z2;
            s3 += qpB[(long)j * kB + b] * vpB[(long)j * kB + b] * z2;
        }
    }
    float a = (s2 >= 0.0f) ? sqrtf(s2) : -sqrtf(-s2);
    float c = (s3 >= 0.0f) ? sqrtf(s3) : -sqrtf(-s3);
    if (o < kO) {
        feat[(long)b * 4000 + baseA + o] = a;
        feat[(long)b * 4000 + baseB + o] = c;
    }
    red[threadIdx.x] = (o < kO) ? a * a : 0.0f; __syncthreads();
    for (int s = 128; s > 0; s >>= 1) {
        if (threadIdx.x < s) red[threadIdx.x] += red[threadIdx.x + s];
        __syncthreads();
    }
    if (threadIdx.x == 0) atomicAdd(&norms[b * 8 + slotA], red[0]);
    __syncthreads();
    red[threadIdx.x] = (o < kO) ? c * c : 0.0f; __syncthreads();
    for (int s = 128; s > 0; s >>= 1) {
        if (threadIdx.x < s) red[threadIdx.x] += red[threadIdx.x + s];
        __syncthreads();
    }
    if (threadIdx.x == 0) atomicAdd(&norms[b * 8 + slotB], red[0]);
}

// bin (row 0) and p2 (rows 1..50): one block per (b,row), 256-thread dot of 4000
__global__ void heads_small_k(const float* __restrict__ feat,
                              const float* __restrict__ binW, const float* __restrict__ binB,
                              const float* __restrict__ p2W,  const float* __restrict__ p2B,
                              float* __restrict__ out)
{
    __shared__ float red[256];
    int blk = blockIdx.x;
    int b = blk / (1 + kTOK), r = blk % (1 + kTOK);
    const float* f = feat + (long)b * 4000;
    const float* w = (r == 0) ? binW : p2W + (long)(r - 1) * 4000;
    float s = 0.0f;
    for (int k = threadIdx.x; k < 4000; k += 256) s += f[k] * w[k];
    red[threadIdx.x] = s; __syncthreads();
    for (int st = 128; st > 0; st >>= 1) {
        if (threadIdx.x < st) red[threadIdx.x] += red[threadIdx.x + st];
        __syncthreads();
    }
    if (threadIdx.x == 0) {
        float v = red[0] + ((r == 0) ? binB[0] : p2B[r - 1]);
        out[(long)b * kOUTC + ((r == 0) ? 0 : (1 + kANS + (r - 1)))] = v;
    }
}

// ============================================================
// Host-side orchestration
// ============================================================
static inline void gemm(hipStream_t s,
                        const float* A, long lda, long aB,
                        const float* Bm, long ldbK, long ldbN, long bB,
                        float* C, long csm, long csn, long cB,
                        const float* bias, int M, int N, int K, int flags, int batch)
{
    dim3 grid((N + 31) / 32, (M + 255) / 256, batch);
    wmma_gemm_f32<<<grid, 256, 0, s>>>(A, lda, aB, Bm, ldbK, ldbN, bB,
                                       C, csm, csn, cB, bias, M, N, K, flags);
}

extern "C" void kernel_launch(void* const* d_in, const int* in_sizes, int n_in,
                              void* d_out, int out_size, void* d_ws, size_t ws_size,
                              hipStream_t stream)
{
    (void)in_sizes; (void)n_in; (void)out_size; (void)ws_size;
    auto cdiv = [](long a, long b) { return (int)((a + b - 1) / b); };

    // ---- inputs ----
    const int*   data  = (const int*)d_in[0];
    const float* img   = (const float*)d_in[1];   // [B,2048,196]
    const float* glove = (const float*)d_in[2];   // [B,T,300]
    /* cvec_token d_in[3] unused */
    const float* tok   = (const float*)d_in[4];   // [B,50,300]
    const float* P[47];
    for (int i = 0; i < 47; ++i) P[i] = (const float*)d_in[5 + i];
    const float *emb=P[0], *Wih=P[1], *Whh=P[2], *bih=P[3], *bhh=P[4];
    const float *conviW=P[5], *conviB=P[6], *convoW=P[7], *convoB=P[8];
    const float *li2W=P[9], *li2B=P[10], *li3W=P[11], *li3B=P[12];
    const float *lo2W=P[13], *lo2B=P[14], *lo3W=P[15], *lo3B=P[16];
    const float *qatt1W=P[17], *qatt1B=P[18], *qatt2W=P[19], *qatt2B=P[20];
    const float *iatt1W=P[21], *iatt1B=P[22], *iatt2W=P[23], *iatt2B=P[24];
    const float *oatt1W=P[25], *oatt1B=P[26], *oatt2W=P[27], *oatt2B=P[28];
    const float *binW=P[29], *binB=P[30], *p1W=P[31], *p1B=P[32], *p2W=P[33], *p2B=P[34];
    const float *qW[6], *qb[6];
    for (int i = 0; i < 6; ++i) { qW[i] = P[35 + 2*i]; qb[i] = P[36 + 2*i]; }
    float* out = (float*)d_out;

    // ---- workspace carve (floats; every size a multiple of 4 -> float4 safe) ----
    float* ws = (float*)d_ws;
    size_t off = 0;
    auto take = [&](size_t n) { float* p = ws + off; off += n; return p; };
    float* bsum = take(4 * kH);
    float* xseq = take((size_t)kT * kB * 600);
    float* hbuf = take((size_t)kB * kH);
    float* cbuf = take((size_t)kB * kH);
    float* gbuf = take((size_t)4 * kH * kB);
    float* hs   = take((size_t)kT * kB * kH);
    float* a1q  = take((size_t)512 * kT * kB);
    float* qlog = take((size_t)kT * kB);
    float* qf   = take((size_t)kB * kH);
    float* qp[6];
    for (int i = 0; i < 6; ++i) qp[i] = take((size_t)kJ * kB);
    float* vpbuf = take((size_t)4 * kJ * kNIMG);          // chunk buffer (reused)
    float* zpi   = take((size_t)kB * kO * kNIMG);
    float* a1i   = take((size_t)kB * 512 * kNIMG);
    float* ilog  = take((size_t)kB * 2 * kNIMG);
    float* ifeat = take((size_t)kB * 2 * kIMGC);
    float* zpo   = take((size_t)kB * kO * kTOK);
    float* a1o   = take((size_t)kB * 512 * kTOK);
    float* olog  = take((size_t)kB * 2 * kTOK);
    float* ofeat = take((size_t)kB * 2 * kTOKE);
    float* vpm[4];
    for (int i = 0; i < 4; ++i) vpm[i] = take((size_t)kJ * kB);
    float* feat  = take((size_t)kB * 4000);
    float* norms = take((size_t)kB * 8);

    // ---- init ----
    zero_k<<<1, 256, 0, stream>>>(norms, kB * 8);
    zero_k<<<cdiv(kB * kH, 256), 256, 0, stream>>>(hbuf, kB * kH);
    zero_k<<<cdiv(kB * kH, 256), 256, 0, stream>>>(cbuf, kB * kH);
    add2_k<<<cdiv(4 * kH, 256), 256, 0, stream>>>(bsum, bih, bhh, 4 * kH);

    // ---- embedding + glove concat ----
    embed_k<<<cdiv((long)kT * kB * 600, 256), 256, 0, stream>>>(data, emb, glove, xseq);

    // ---- LSTM over T steps: g = Wih@x_t^T + Whh@h^T + bsum ----
    for (int t = 0; t < kT; ++t) {
        gemm(stream, Wih, 600, 0,
             xseq + (size_t)t * kB * 600, 1, 600, 0,
             gbuf, kB, 1, 0, bsum, 4 * kH, kB, 600, 0, 1);
        gemm(stream, Whh, kH, 0,
             hbuf, 1, kH, 0,
             gbuf, kB, 1, 0, nullptr, 4 * kH, kB, kH, GEMM_ACC, 1);
        lstm_cell_k<<<cdiv(kB * kH, 256), 256, 0, stream>>>(gbuf, hbuf, cbuf,
                                                            hs + (size_t)t * kB * kH);
    }

    // ---- question attention ----
    gemm(stream, qatt1W, kH, 0, hs, 1, kH, 0,
         a1q, kT * kB, 1, 0, qatt1B, 512, kT * kB, kH, GEMM_RELU, 1);
    qatt2_k<<<cdiv(kT * kB, 256), 256, 0, stream>>>(a1q, qatt2W, qatt2B, qlog);
    q_softmax_k<<<1, 32, 0, stream>>>(qlog);
    qfeat_k<<<cdiv(kB * kH, 256), 256, 0, stream>>>(qlog, hs, qf);

    // ---- six question projections qp[i] = qW[i] @ qf^T + qb[i]  [5000,32] ----
    for (int i = 0; i < 6; ++i)
        gemm(stream, qW[i], kH, 0, qf, 1, kH, 0,
             qp[i], kB, 1, 0, qb[i], kJ, kB, kH, 0, 1);

    // ---- image MFB: vp chunks of 4 batch, fused pool ----
    for (int b0 = 0; b0 < kB; b0 += 4) {
        gemm(stream, conviW, kIMGC, 0,
             img + (size_t)b0 * kIMGC * kNIMG, kNIMG, 1, (long)kIMGC * kNIMG,
             vpbuf, kNIMG, 1, (long)kJ * kNIMG,
             conviB, kJ, kNIMG, kIMGC, 0, 4);
        mfb_pool_k<<<cdiv(4L * kO * kNIMG, 256), 256, 0, stream>>>(
            qp[0], vpbuf, zpi, b0, 4, kNIMG, (long)kJ * kNIMG);
    }
    {
        dim3 g(cdiv(kO * kNIMG, 256), kB);
        ssqrt_norm_k<<<g, 256, 0, stream>>>(zpi, kO * kNIMG, (long)kO * kNIMG, 0, norms);
        l2div_k<<<g, 256, 0, stream>>>(zpi, kO * kNIMG, (long)kO * kNIMG, 0, norms);
    }
    // iatt1 (batched over b via grid.z)
    gemm(stream, iatt1W, kO, 0,
         zpi, kNIMG, 1, (long)kO * kNIMG,
         a1i, kNIMG, 1, (long)512 * kNIMG,
         iatt1B, 512, kNIMG, kO, GEMM_RELU, kB);
    att2_k<<<cdiv(kB * 2 * kNIMG, 256), 256, 0, stream>>>(a1i, (long)512 * kNIMG,
                                                          iatt2W, iatt2B, ilog, 2, kNIMG);
    softmax_loc_k<<<kB * 2, 256, 0, stream>>>(ilog, kNIMG);
    att_wsum_k<<<cdiv((long)kB * 2 * kIMGC, 256), 256, 0, stream>>>(
        ilog, img, ifeat, 2, kIMGC, kNIMG, (long)kIMGC * kNIMG, kNIMG, 1);

    // ---- OCR-token MFB ----
    for (int b0 = 0; b0 < kB; b0 += 4) {
        gemm(stream, convoW, kTOKE, 0,
             tok + (size_t)b0 * kTOK * kTOKE, 1, kTOKE, (long)kTOK * kTOKE,
             vpbuf, kTOK, 1, (long)kJ * kTOK,
             convoB, kJ, kTOK, kTOKE, 0, 4);
        mfb_pool_k<<<cdiv(4L * kO * kTOK, 256), 256, 0, stream>>>(
            qp[3], vpbuf, zpo, b0, 4, kTOK, (long)kJ * kTOK);
    }
    {
        dim3 g(cdiv(kO * kTOK, 256), kB);
        ssqrt_norm_k<<<g, 256, 0, stream>>>(zpo, kO * kTOK, (long)kO * kTOK, 1, norms);
        l2div_k<<<g, 256, 0, stream>>>(zpo, kO * kTOK, (long)kO * kTOK, 1, norms);
    }
    gemm(stream, oatt1W, kO, 0,
         zpo, kTOK, 1, (long)kO * kTOK,
         a1o, kTOK, 1, (long)512 * kTOK,
         oatt1B, 512, kTOK, kO, GEMM_RELU, kB);
    att2_k<<<cdiv(kB * 2 * kTOK, 256), 256, 0, stream>>>(a1o, (long)512 * kTOK,
                                                         oatt2W, oatt2B, olog, 2, kTOK);
    softmax_loc_k<<<kB * 2, 256, 0, stream>>>(olog, kTOK);
    att_wsum_k<<<cdiv((long)kB * 2 * kTOKE, 256), 256, 0, stream>>>(
        olog, tok, ofeat, 2, kTOKE, kTOK, (long)kTOK * kTOKE, 1, kTOKE);

    // ---- MFH (image branch): vp2 = li2@i_feat^T, vp3 = li3@i_feat^T ----
    gemm(stream, li2W, 2 * kIMGC, 0, ifeat, 1, 2 * kIMGC, 0,
         vpm[0], kB, 1, 0, li2B, kJ, kB, 2 * kIMGC, 0, 1);
    gemm(stream, li3W, 2 * kIMGC, 0, ifeat, 1, 2 * kIMGC, 0,
         vpm[1], kB, 1, 0, li3B, kJ, kB, 2 * kIMGC, 0, 1);
    {
        dim3 g(cdiv(kO, 256), kB);
        mfh_pool_k<<<g, 256, 0, stream>>>(qp[1], vpm[0], qp[2], vpm[1],
                                          feat, 0, kO, 2, 3, norms);
        l2div_k<<<g, 256, 0, stream>>>(feat + 0,  kO, 4000, 2, norms);
        l2div_k<<<g, 256, 0, stream>>>(feat + kO, kO, 4000, 3, norms);
    }

    // ---- MFH (OCR branch): vp5 = lo2@o_feat^T, vp6 = lo3@o_feat^T ----
    gemm(stream, lo2W, 2 * kTOKE, 0, ofeat, 1, 2 * kTOKE, 0,
         vpm[2], kB, 1, 0, lo2B, kJ, kB, 2 * kTOKE, 0, 1);
    gemm(stream, lo3W, 2 * kTOKE, 0, ofeat, 1, 2 * kTOKE, 0,
         vpm[3], kB, 1, 0, lo3B, kJ, kB, 2 * kTOKE, 0, 1);
    {
        dim3 g(cdiv(kO, 256), kB);
        mfh_pool_k<<<g, 256, 0, stream>>>(qp[4], vpm[2], qp[5], vpm[3],
                                          feat, 2 * kO, 3 * kO, 4, 5, norms);
        l2div_k<<<g, 256, 0, stream>>>(feat + 2 * kO, kO, 4000, 4, norms);
        l2div_k<<<g, 256, 0, stream>>>(feat + 3 * kO, kO, 4000, 5, norms);
    }

    // ---- heads: pred1 via WMMA GEMM (writes transposed straight into d_out) ----
    gemm(stream, p1W, 4000, 0, feat, 1, 4000, 0,
         out + 1, 1, kOUTC, 0, p1B, kANS, kB, 4000, 0, 1);
    heads_small_k<<<kB * (1 + kTOK), 256, 0, stream>>>(feat, binW, binB, p2W, p2B, out);
}
